// LocalModel_76527727280750
// MI455X (gfx1250) — compile-verified, matrix operand order checked by reading, not run on hardware
//
#include <hip/hip_runtime.h>
#include <hip/hip_bf16.h>

#define B_   2
#define S_   2048
#define E_   512
#define H_   8
#define DH_  64
#define W_   5
#define PAD_ 2
#define SK_  (S_ - W_ + 1)   /* 2044 */
#define SKP_ 2048            /* padded key rows */
#define HID_ 2048
#define OUT_ 6
#define L_   6
#define FEAT_ (DH_ * W_)     /* 320 */
#define FEATP_ 328           /* LDS row stride for K slab (bank-conflict pad) */
#define VROW_  40            /* LDS row stride for V slab / P buffer */
#define SCALE_ 0.125f        /* 64^-0.5 */
#define AT_WAVES 8

typedef __bf16 bf16;
typedef __attribute__((ext_vector_type(16))) __bf16 v16bf;
typedef __attribute__((ext_vector_type(8)))  float  v8f;
typedef __attribute__((ext_vector_type(4)))  float  f32x4;
typedef __attribute__((ext_vector_type(4)))  int    i32x4;

// AS-qualified int4 pointee types matching the async-LDS builtin signature
// (clang HIP prints AS1 as __device__, AS3 as __shared__).
typedef __attribute__((address_space(1))) i32x4 gl_i32x4;
typedef __attribute__((address_space(3))) i32x4 ds_i32x4;

// ---------------------------------------------------------------------------
// Fragment loader: ISA 16-bit A-matrix 16x32 layout (and BT rows for B):
// two contiguous 16-byte chunks per lane at +0 and +16 bf16.
// ---------------------------------------------------------------------------
__device__ __forceinline__ v16bf load_frag(const bf16* p) {
  union { v16bf v; f32x4 q[2]; } u;
  u.q[0] = *(const f32x4*)p;
  u.q[1] = *(const f32x4*)(p + 16);
  return u.v;
}

__device__ __forceinline__ v8f wmma_bf16(v16bf a, v16bf b, v8f c) {
  return __builtin_amdgcn_wmma_f32_16x16x32_bf16(false, a, false, b,
                                                 (short)0, c, false, false);
}

// ---------------------------------------------------------------------------
// 16-byte global->LDS copy: async (ASYNCcnt) if the toolchain has the builtin,
// else synchronous load+ds_store fallback. Guarded so host pass also compiles.
// ---------------------------------------------------------------------------
#if defined(__AMDGCN__) && __has_builtin(__builtin_amdgcn_global_load_async_to_lds_b128)
#define ASYNC_LDS 1
__device__ __forceinline__ void cp16_g2l(const bf16* g, bf16* l) {
  __builtin_amdgcn_global_load_async_to_lds_b128(
      (gl_i32x4*)g, (ds_i32x4*)l, 0, 0);
}
__device__ __forceinline__ void cp_wait() {
#if __has_builtin(__builtin_amdgcn_s_wait_asynccnt)
  __builtin_amdgcn_s_wait_asynccnt(0);
#else
  asm volatile("s_wait_asynccnt 0" ::: "memory");
#endif
}
#else
#define ASYNC_LDS 0
__device__ __forceinline__ void cp16_g2l(const bf16* g, bf16* l) {
  *(f32x4*)l = *(const f32x4*)g;
}
__device__ __forceinline__ void cp_wait() {}
#endif

// ---------------------------------------------------------------------------
// Weight convert + transpose: src f32 [K,N] -> dst bf16 [N,K]
// ---------------------------------------------------------------------------
__global__ void convtr_kernel(const float* __restrict__ src, bf16* __restrict__ dst,
                              int K, int N) {
  int idx = blockIdx.x * blockDim.x + threadIdx.x;
  if (idx >= K * N) return;
  int kk = idx % K;
  int n  = idx / K;
  dst[(size_t)n * K + kk] = (bf16)src[(size_t)kk * N + n];
}

// ---------------------------------------------------------------------------
// Embedding + sinusoidal positional encoding
// ---------------------------------------------------------------------------
__global__ void embed_kernel(const int* __restrict__ inputs, const float* __restrict__ emb,
                             float* __restrict__ x, bf16* __restrict__ xb) {
  int idx = blockIdx.x * blockDim.x + threadIdx.x;
  if (idx >= B_ * S_ * E_) return;
  int e = idx % E_;
  int s = (idx / E_) % S_;
  int b = idx / (E_ * S_);
  int tok = inputs[b * S_ + s];
  float ang = (float)s * __expf(-(float)(e & ~1) * (9.210340371976184f / (float)E_));
  float pe  = (e & 1) ? __cosf(ang) : __sinf(ang);
  float val = emb[(size_t)tok * E_ + e] + pe;
  x[idx]  = val;
  xb[idx] = (bf16)val;
}

// ---------------------------------------------------------------------------
// Generic tiled bf16 WMMA GEMM: C[M,N] = A[M,K] * BT[N,K]^T + bias (+epilogue)
// block = 256 threads = 8 waves; wave tile 16x64; block tile 64x128.
// MODE 0: f32 out; MODE 1: relu -> bf16 out; MODE 2: + residual -> f32 out
// ---------------------------------------------------------------------------
template<int MODE>
__global__ __launch_bounds__(256)
void gemm_kernel(const bf16* __restrict__ A, const bf16* __restrict__ BT,
                 const float* __restrict__ bias, const float* __restrict__ resid,
                 float* __restrict__ outf, bf16* __restrict__ outb,
                 int M, int N, int K) {
  int lane = threadIdx.x & 31;
  int wave = threadIdx.x >> 5;
  int wm = wave & 3, wn = wave >> 2;
  int l16 = lane & 15, half = lane >> 4;
  int row0 = blockIdx.x * 64  + wm * 16;
  int col0 = blockIdx.y * 128 + wn * 64;

  const bf16* ap  = A  + (size_t)(row0 + l16) * K + half * 8;
  const bf16* bp0 = BT + (size_t)(col0 +  0 + l16) * K + half * 8;
  const bf16* bp1 = BT + (size_t)(col0 + 16 + l16) * K + half * 8;
  const bf16* bp2 = BT + (size_t)(col0 + 32 + l16) * K + half * 8;
  const bf16* bp3 = BT + (size_t)(col0 + 48 + l16) * K + half * 8;

  v8f zero = {0.f,0.f,0.f,0.f,0.f,0.f,0.f,0.f};
  v8f acc0 = zero, acc1 = zero, acc2 = zero, acc3 = zero;

  for (int kk = 0; kk < K; kk += 32) {
    v16bf a = load_frag(ap + kk);
    if (kk + 128 < K) __builtin_prefetch(ap + kk + 128, 0, 1);  // global_prefetch_b8
    acc0 = wmma_bf16(a, load_frag(bp0 + kk), acc0);
    acc1 = wmma_bf16(a, load_frag(bp1 + kk), acc1);
    acc2 = wmma_bf16(a, load_frag(bp2 + kk), acc2);
    acc3 = wmma_bf16(a, load_frag(bp3 + kk), acc3);
  }

  float bv0 = bias[col0 +  0 + l16];
  float bv1 = bias[col0 + 16 + l16];
  float bv2 = bias[col0 + 32 + l16];
  float bv3 = bias[col0 + 48 + l16];

#pragma unroll
  for (int v = 0; v < 8; ++v) {
    int row = row0 + v + 8 * half;
    size_t o0 = (size_t)row * N + col0 + l16;
    float c0 = acc0[v] + bv0, c1 = acc1[v] + bv1, c2 = acc2[v] + bv2, c3 = acc3[v] + bv3;
    if (MODE == 0) {
      outf[o0] = c0; outf[o0+16] = c1; outf[o0+32] = c2; outf[o0+48] = c3;
    } else if (MODE == 1) {
      outb[o0]    = (bf16)fmaxf(c0, 0.f); outb[o0+16] = (bf16)fmaxf(c1, 0.f);
      outb[o0+32] = (bf16)fmaxf(c2, 0.f); outb[o0+48] = (bf16)fmaxf(c3, 0.f);
    } else {
      outf[o0]    = c0 + resid[o0];    outf[o0+16] = c1 + resid[o0+16];
      outf[o0+32] = c2 + resid[o0+32]; outf[o0+48] = c3 + resid[o0+48];
    }
  }
}

// ---------------------------------------------------------------------------
// Unfold: build q_un (scaled, zero-padded window), k_un (padded to SKP rows),
// and v_sumT = transposed window-sum of v:  [B,H,DH,SKP]
// ---------------------------------------------------------------------------
__global__ void unfold_kernel(const float* __restrict__ q, const float* __restrict__ k,
                              const float* __restrict__ v,
                              bf16* __restrict__ q_un, bf16* __restrict__ k_un,
                              bf16* __restrict__ v_sumT) {
  int idx = blockIdx.x * blockDim.x + threadIdx.x;
  if (idx >= B_ * H_ * S_ * DH_) return;
  int d = idx % DH_;
  int s = (idx / DH_) % S_;
  int h = (idx / (DH_ * S_)) % H_;
  int b = idx / (DH_ * S_ * H_);
  int bh = b * H_ + h;
  int ecol = h * DH_ + d;

  bf16* qo = q_un + ((size_t)bh * S_ + s) * FEAT_ + d * W_;
#pragma unroll
  for (int j = 0; j < W_; ++j) {
    int sr = s + j - PAD_;
    float val = (sr >= 0 && sr < S_) ? q[((size_t)b * S_ + sr) * E_ + ecol] * SCALE_ : 0.f;
    qo[j] = (bf16)val;
  }

  bf16* ko = k_un + ((size_t)bh * SKP_ + s) * FEAT_ + d * W_;
  float vs = 0.f;
  if (s < SK_) {
#pragma unroll
    for (int j = 0; j < W_; ++j) {
      ko[j] = (bf16)k[((size_t)b * S_ + s + j) * E_ + ecol];
      vs   += v[((size_t)b * S_ + s + j) * E_ + ecol];
    }
  } else {
#pragma unroll
    for (int j = 0; j < W_; ++j) ko[j] = (bf16)0.f;
  }
  v_sumT[((size_t)bh * DH_ + d) * SKP_ + s] = (bf16)vs;
}

// ---------------------------------------------------------------------------
// Fused flash-style attention, 8 waves/block sharing K/V chunks through LDS.
// Per 32-key chunk: async-stage next chunk's K-slab (32x320) + V-slab (64x32)
// into LDS (double buffered, ASYNCcnt), 20 score WMMAs from LDS, online
// softmax, P transposed through LDS, 4 PV WMMAs. Bank-conflict-free strides.
// ---------------------------------------------------------------------------
__global__ __launch_bounds__(256)
void attn_kernel(const bf16* __restrict__ q_un, const bf16* __restrict__ k_un,
                 const bf16* __restrict__ v_sumT, float* __restrict__ attn_out) {
  __shared__ __align__(16) bf16 kstage[2][32 * FEATP_];   // 2 x 20.5 KB
  __shared__ __align__(16) bf16 vstage[2][64 * VROW_];    // 2 x 5 KB
  __shared__ __align__(16) bf16 plds[AT_WAVES][16 * VROW_];

  int tid  = threadIdx.x;
  int lane = tid & 31, wave = tid >> 5;
  int l16 = lane & 15, half = lane >> 4;
  int bh = blockIdx.y;
  int b = bh / H_, h = bh % H_;
  int q0 = (blockIdx.x * AT_WAVES + wave) * 16;

  const bf16* qu = q_un   + (size_t)bh * S_   * FEAT_;
  const bf16* ku = k_un   + (size_t)bh * SKP_ * FEAT_;
  const bf16* vT = v_sumT + (size_t)bh * DH_  * SKP_;

  // stage one 32-key chunk: 1280 16B units of K + 256 16B units of V
  auto stage = [&](int bi, int k0) {
#pragma unroll
    for (int i = 0; i < 6; ++i) {
      int u = tid + i * 256;
      if (u < 1280) {
        int key = u / 40, off = (u % 40) * 8;
        cp16_g2l(ku + (size_t)(k0 + key) * FEAT_ + off,
                 &kstage[bi][key * FEATP_ + off]);
      } else {
        int u2 = u - 1280;
        int dh = u2 >> 2, off = (u2 & 3) * 8;
        cp16_g2l(vT + (size_t)dh * SKP_ + k0 + off,
                 &vstage[bi][dh * VROW_ + off]);
      }
    }
  };

  // cache the 10 Q fragments (16 rows x 320 feats) in registers
  v16bf aq[10];
  const bf16* aqp = qu + (size_t)(q0 + l16) * FEAT_ + half * 8;
#pragma unroll
  for (int f = 0; f < 10; ++f) aq[f] = load_frag(aqp + f * 32);

  v8f zero = {0.f,0.f,0.f,0.f,0.f,0.f,0.f,0.f};
  v8f o0 = zero, o1 = zero, o2 = zero, o3 = zero;
  float m[8], l[8];
#pragma unroll
  for (int v = 0; v < 8; ++v) { m[v] = -3.0e38f; l[v] = 0.f; }

  bf16* pl = plds[wave];

  stage(0, 0);
  cp_wait();
  __syncthreads();

  int cur = 0;
  for (int k0 = 0; k0 < SKP_; k0 += 32) {
    if (k0 + 32 < SKP_) stage(cur ^ 1, k0 + 32);

    const bf16* kb = kstage[cur];
    const bf16* vb = vstage[cur];

    v8f s0 = zero, s1 = zero;
    const bf16* b0p = kb + (size_t)l16        * FEATP_ + half * 8;
    const bf16* b1p = kb + (size_t)(16 + l16) * FEATP_ + half * 8;
#pragma unroll
    for (int f = 0; f < 10; ++f) {
      s0 = wmma_bf16(aq[f], load_frag(b0p + f * 32), s0);
      s1 = wmma_bf16(aq[f], load_frag(b1p + f * 32), s1);
    }
    // mask padded key columns (>= SK)
    if (k0 + l16 >= SK_) {
      for (int v = 0; v < 8; ++v) s0[v] = -1e30f;
    }
    if (k0 + 16 + l16 >= SK_) {
      for (int v = 0; v < 8; ++v) s1[v] = -1e30f;
    }
    // online softmax; xor masks < 16 keep the two half-wave row groups apart
    float alpha[8];
#pragma unroll
    for (int v = 0; v < 8; ++v) {
      float mx = fmaxf(s0[v], s1[v]);
      mx = fmaxf(mx, __shfl_xor(mx, 1));
      mx = fmaxf(mx, __shfl_xor(mx, 2));
      mx = fmaxf(mx, __shfl_xor(mx, 4));
      mx = fmaxf(mx, __shfl_xor(mx, 8));
      float mnew = fmaxf(m[v], mx);
      float a_  = __expf(m[v] - mnew);
      float p0  = __expf(s0[v] - mnew);
      float p1  = __expf(s1[v] - mnew);
      float rs  = p0 + p1;
      rs += __shfl_xor(rs, 1); rs += __shfl_xor(rs, 2);
      rs += __shfl_xor(rs, 4); rs += __shfl_xor(rs, 8);
      l[v] = l[v] * a_ + rs;
      m[v] = mnew;
      alpha[v] = a_;
      s0[v] = p0; s1[v] = p1;
    }
#pragma unroll
    for (int v = 0; v < 8; ++v) {
      o0[v] *= alpha[v]; o1[v] *= alpha[v]; o2[v] *= alpha[v]; o3[v] *= alpha[v];
    }
    // P (C-layout) -> LDS -> reload as A fragment (16x32 bf16)
#pragma unroll
    for (int v = 0; v < 8; ++v) {
      int r = v + 8 * half;
      pl[r * VROW_ +      l16] = (bf16)s0[v];
      pl[r * VROW_ + 16 + l16] = (bf16)s1[v];
    }
    asm volatile("s_wait_dscnt 0" ::: "memory");
    v16bf pf = load_frag(pl + l16 * VROW_ + half * 8);
    const bf16* v0p = vb + (size_t)l16 * VROW_ + half * 8;
    o0 = wmma_bf16(pf, load_frag(v0p             ), o0);
    o1 = wmma_bf16(pf, load_frag(v0p + 16 * VROW_), o1);
    o2 = wmma_bf16(pf, load_frag(v0p + 32 * VROW_), o2);
    o3 = wmma_bf16(pf, load_frag(v0p + 48 * VROW_), o3);

    __syncthreads();   // all waves done reading cur slab
    cp_wait();         // this wave's prefetch into the other slab landed
    __syncthreads();   // all waves' prefetches landed
    cur ^= 1;
  }

#pragma unroll
  for (int v = 0; v < 8; ++v) {
    float inv = 1.f / l[v];
    int row = q0 + v + 8 * half;
    float* orow = attn_out + ((size_t)(b * S_ + row)) * E_ + h * DH_;
    orow[     l16] = o0[v] * inv;
    orow[16 + l16] = o1[v] * inv;
    orow[32 + l16] = o2[v] * inv;
    orow[48 + l16] = o3[v] * inv;
  }
}

// ---------------------------------------------------------------------------
// LayerNorm over E=512 per row; one wave per row (16 elems/lane, wave32).
// ---------------------------------------------------------------------------
__global__ __launch_bounds__(256)
void ln_kernel(const float* __restrict__ in, const float* __restrict__ g,
               const float* __restrict__ bta, float* __restrict__ outf,
               bf16* __restrict__ outb) {
  int wave = threadIdx.x >> 5, lane = threadIdx.x & 31;
  int row = blockIdx.x * 8 + wave;
  const float* r = in + (size_t)row * E_;
  float vals[16];
  float s = 0.f, s2 = 0.f;
#pragma unroll
  for (int i = 0; i < 16; ++i) {
    float x = r[lane + 32 * i];
    vals[i] = x; s += x; s2 += x * x;
  }
  for (int off = 1; off < 32; off <<= 1) { s += __shfl_xor(s, off); s2 += __shfl_xor(s2, off); }
  float mean = s * (1.f / E_);
  float var  = s2 * (1.f / E_) - mean * mean;
  float rstd = rsqrtf(var + 1e-5f);
  float* of  = outf + (size_t)row * E_;
  bf16* obp  = outb + (size_t)row * E_;
#pragma unroll
  for (int i = 0; i < 16; ++i) {
    int e = lane + 32 * i;
    float y = (vals[i] - mean) * rstd * g[e] + bta[e];
    of[e]  = y;
    obp[e] = (bf16)y;
  }
}

// ---------------------------------------------------------------------------
// Final head: out[b,o] = x[b,:] . out_w[:,o] + out_b[o]   (12 outputs)
// ---------------------------------------------------------------------------
__global__ __launch_bounds__(256)
void out_head_kernel(const float* __restrict__ x, const float* __restrict__ ow,
                     const float* __restrict__ ob, float* __restrict__ out) {
  __shared__ float red[256];
  int b = blockIdx.x / OUT_, o = blockIdx.x % OUT_;
  const float* xr = x + (size_t)b * (S_ * E_);
  float acc = 0.f;
  for (int f = threadIdx.x; f < S_ * E_; f += 256)
    acc += xr[f] * ow[(size_t)f * OUT_ + o];
  red[threadIdx.x] = acc;
  __syncthreads();
  for (int st = 128; st > 0; st >>= 1) {
    if ((int)threadIdx.x < st) red[threadIdx.x] += red[threadIdx.x + st];
    __syncthreads();
  }
  if (threadIdx.x == 0) out[b * OUT_ + o] = red[0] + ob[o];
}

// ---------------------------------------------------------------------------
extern "C" void kernel_launch(void* const* d_in, const int* in_sizes, int n_in,
                              void* d_out, int out_size, void* d_ws, size_t ws_size,
                              hipStream_t stream) {
  (void)in_sizes; (void)n_in; (void)out_size; (void)ws_size;
  const int*   inputs = (const int*)  d_in[0];
  const float* emb    = (const float*)d_in[1];
  const float* ln_w   = (const float*)d_in[2];
  const float* ln_b   = (const float*)d_in[3];
  const float* q_w  = (const float*)d_in[4];  const float* q_b  = (const float*)d_in[5];
  const float* k_w  = (const float*)d_in[6];  const float* k_b  = (const float*)d_in[7];
  const float* v_w  = (const float*)d_in[8];  const float* v_b  = (const float*)d_in[9];
  const float* fc1w = (const float*)d_in[10]; const float* fc1b = (const float*)d_in[11];
  const float* fc2w = (const float*)d_in[12]; const float* fc2b = (const float*)d_in[13];
  const float* outw = (const float*)d_in[14]; const float* outb_ = (const float*)d_in[15];

  char* ws = (char*)d_ws;
  size_t off = 0;
  auto alloc = [&](size_t bytes) -> void* {
    void* p = ws + off;
    off = (off + bytes + 255) & ~(size_t)255;
    return p;
  };

  const size_t NROW = (size_t)B_ * S_;          // 4096
  float* x       = (float*)alloc(NROW * E_ * 4);
  bf16*  xb      = (bf16*) alloc(NROW * E_ * 2);
  float* qbuf    = (float*)alloc(NROW * E_ * 4);
  float* kbuf    = (float*)alloc(NROW * E_ * 4);
  float* vbuf    = (float*)alloc(NROW * E_ * 4);
  float* att     = (float*)alloc(NROW * E_ * 4);
  float* x1      = (float*)alloc(NROW * E_ * 4);
  bf16*  x1b     = (bf16*) alloc(NROW * E_ * 2);
  bf16*  hbuf    = (bf16*) alloc(NROW * HID_ * 2);
  float* x2      = (float*)alloc(NROW * E_ * 4);
  bf16*  q_un    = (bf16*) alloc((size_t)B_ * H_ * S_   * FEAT_ * 2);
  bf16*  k_un    = (bf16*) alloc((size_t)B_ * H_ * SKP_ * FEAT_ * 2);
  bf16*  v_sumT  = (bf16*) alloc((size_t)B_ * H_ * DH_  * SKP_  * 2);
  bf16*  qwT     = (bf16*) alloc((size_t)E_ * E_ * 2);
  bf16*  kwT     = (bf16*) alloc((size_t)E_ * E_ * 2);
  bf16*  vwT     = (bf16*) alloc((size_t)E_ * E_ * 2);
  bf16*  fc1T    = (bf16*) alloc((size_t)HID_ * E_ * 2);
  bf16*  fc2T    = (bf16*) alloc((size_t)E_ * HID_ * 2);

  // --- weight convert + transpose (bf16 [N,K] layout for WMMA B operand) ---
  int nEE = E_ * E_, nEH = E_ * HID_;
  convtr_kernel<<<(nEE + 255) / 256, 256, 0, stream>>>(q_w,  qwT,  E_,   E_);
  convtr_kernel<<<(nEE + 255) / 256, 256, 0, stream>>>(k_w,  kwT,  E_,   E_);
  convtr_kernel<<<(nEE + 255) / 256, 256, 0, stream>>>(v_w,  vwT,  E_,   E_);
  convtr_kernel<<<(nEH + 255) / 256, 256, 0, stream>>>(fc1w, fc1T, E_,   HID_);
  convtr_kernel<<<(nEH + 255) / 256, 256, 0, stream>>>(fc2w, fc2T, HID_, E_);

  // --- embedding + positional encoding ---
  int nXE = B_ * S_ * E_;
  embed_kernel<<<(nXE + 255) / 256, 256, 0, stream>>>(inputs, emb, x, xb);

  // --- transformer layers ---
  for (int layer = 0; layer < L_; ++layer) {
    dim3 gQKV(64, 4);   // M=4096/64, N=512/128
    gemm_kernel<0><<<gQKV, 256, 0, stream>>>(xb, qwT, q_b, nullptr, qbuf, nullptr, 4096, E_, E_);
    gemm_kernel<0><<<gQKV, 256, 0, stream>>>(xb, kwT, k_b, nullptr, kbuf, nullptr, 4096, E_, E_);
    gemm_kernel<0><<<gQKV, 256, 0, stream>>>(xb, vwT, v_b, nullptr, vbuf, nullptr, 4096, E_, E_);

    int nUF = B_ * H_ * S_ * DH_;
    unfold_kernel<<<(nUF + 255) / 256, 256, 0, stream>>>(qbuf, kbuf, vbuf, q_un, k_un, v_sumT);

    attn_kernel<<<dim3(S_ / (16 * AT_WAVES), B_ * H_), 32 * AT_WAVES, 0, stream>>>(
        q_un, k_un, v_sumT, att);

    ln_kernel<<<(int)(NROW / 8), 256, 0, stream>>>(att, ln_w, ln_b, x1, x1b);

    gemm_kernel<1><<<dim3(64, 16), 256, 0, stream>>>(x1b, fc1T, fc1b, nullptr, nullptr, hbuf, 4096, HID_, E_);
    gemm_kernel<2><<<dim3(64, 4),  256, 0, stream>>>(hbuf, fc2T, fc2b, x1, x2, nullptr, 4096, E_, HID_);

    ln_kernel<<<(int)(NROW / 8), 256, 0, stream>>>(x2, ln_w, ln_b, x, xb);
  }

  // --- output head ---
  out_head_kernel<<<B_ * OUT_, 256, 0, stream>>>(x, outw, outb_, (float*)d_out);
}
